// CustomConv2d_27771258536556
// MI455X (gfx1250) — matrix operand order repaired and checked by reading, not compiled, not run on hardware
//
#include <hip/hip_runtime.h>

// ---------------------------------------------------------------------------
// Implicit-GEMM 3x3 conv, MI455X (gfx1250, wave32, WMMA), software-pipelined.
// GEMM view: M=256 (co), N=100352 (b*oh*ow), K=1152 (ci*9+kh*3+kw).
// fp32 split into bf16 hi+lo (Markidis): 3x v_wmma_f32_16x16x32_bf16 per tile.
// Splits precomputed into d_ws (x packed (lo<<16)|hi dwords, w hi/lo planes);
// in-kernel-convert fallback if ws is too small.
// Pipeline: B gathers are register-prefetched one K-step ahead (8 VGPRs);
// A (dense weight stream) is cache-primed one K-step ahead with
// global_prefetch (__builtin_prefetch) -> zero VGPR cost, no scratch spills.
// ---------------------------------------------------------------------------

typedef __attribute__((ext_vector_type(16))) __bf16        v16bf;
typedef __attribute__((ext_vector_type(8)))  float         v8f;
typedef __attribute__((ext_vector_type(8)))  unsigned int  v8u;

using u32 = unsigned int;
using u16 = unsigned short;

// Problem constants
constexpr int BATCH = 32, CIN = 128, COUT = 256, HIMG = 56, WIMG = 56;
constexpr int HW   = HIMG * WIMG;      // 3136
constexpr int KTOT = CIN * 9;          // 1152
constexpr int NTOT = BATCH * HW;       // 100352
constexpr int NX   = BATCH * CIN * HW; // 12845056 x elements
constexpr int NW   = COUT * KTOT;      // 294912   w elements

// Tiling: 256 threads = 8 waves; macro-tile 256(M) x 64(N), K step 32.
// Wave grid 4(M) x 2(N): each wave owns 64x32 = 4x2 tiles -> 24 WMMA/step.
constexpr int BM = 256, BN = 64, BK = 32;
constexpr int ASTR = 40;               // halves per LDS row (80 B, 16B-aligned)
constexpr int BSTR = 40;

static __device__ __forceinline__ u32 bf16_rne(u32 u) {
    return (u + 0x7FFFu + ((u >> 16) & 1u)) >> 16;
}
// (lo<<16)|hi packed bf16 split of one f32
static __device__ __forceinline__ u32 split_pack(float f) {
    u32 u = __float_as_uint(f);
    u32 h = bf16_rne(u);
    float r = f - __uint_as_float(h << 16);
    u32 l = bf16_rne(__float_as_uint(r));
    return (l << 16) | h;
}

// A-fragment, 16-bit A 16x32 layout (ISA 7.12.2):
//   lanes 0-15: dwords {0..3, 8..11} of the K-row; lanes 16-31: {4..7,12..15}
static __device__ __forceinline__ v16bf frag_A(const u16* rowp, int half) {
    const u32* p = (const u32*)rowp;
    const int b = half * 4;
    v8u u;
#pragma unroll
    for (int d = 0; d < 4; ++d) { u[d] = p[b + d]; u[4 + d] = p[b + 8 + d]; }
    return __builtin_bit_cast(v16bf, u);
}
// B-fragment, 16-bit B 32x16 layout; tile stored transposed [n][k] in LDS.
static __device__ __forceinline__ v16bf frag_B(const u16* rowp, int half) {
    const u32* p = (const u32*)rowp;
    const int b = half * 4;
    v8u u;
#pragma unroll
    for (int d = 0; d < 4; ++d) { u[d] = p[b + d]; u[4 + d] = p[b + 8 + d]; }
    return __builtin_bit_cast(v16bf, u);
}

// ---------------- pre-pass: split fp32 -> bf16 hi/lo ------------------------
__global__ __launch_bounds__(256)
void splitpack_x(const float4* __restrict__ x, uint4* __restrict__ xpk) {
    const int i = blockIdx.x * 256 + threadIdx.x;
    const float4 v = x[i];
    uint4 o;
    o.x = split_pack(v.x); o.y = split_pack(v.y);
    o.z = split_pack(v.z); o.w = split_pack(v.w);
    xpk[i] = o;
}
__global__ __launch_bounds__(256)
void split_w(const float* __restrict__ w, u16* __restrict__ wh, u16* __restrict__ wl) {
    const int i = blockIdx.x * 256 + threadIdx.x;
    const u32 p = split_pack(w[i]);
    wh[i] = (u16)p;
    wl[i] = (u16)(p >> 16);
}

// ---------------- main GEMM ------------------------------------------------
template <bool PRESPLIT>
__global__ __launch_bounds__(256, 1)
void conv_gemm(const float* __restrict__ xf, const float* __restrict__ wf,
               const u32* __restrict__ xpk, const u16* __restrict__ wh,
               const u16* __restrict__ wl, const float* __restrict__ bias,
               float* __restrict__ out)
{
    __shared__ u16 As_hi[BM * ASTR];
    __shared__ u16 As_lo[BM * ASTR];
    __shared__ u16 Bs_hi[BN * BSTR];
    __shared__ u16 Bs_lo[BN * BSTR];

    const int tid   = threadIdx.x;
    const int lane  = tid & 31;
    const int wave  = tid >> 5;        // 0..7
    const int waveM = wave >> 1;       // 0..3 -> 64-row M band
    const int waveN = wave & 1;        // 0..1 -> 32-col N band
    const int lhalf = lane >> 4;
    const int l16   = lane & 15;
    const int n0    = blockIdx.x * BN;

    // Loop-invariant im2col decode: thread stages k-row `krow` for 8 n-cols.
    const int krow = tid >> 3;          // 0..31
    const int nc0  = (tid & 7) * 8;     // 0..56
    int xbase[8];
    u32 ohow[8];
#pragma unroll
    for (int i = 0; i < 8; ++i) {
        const int n  = n0 + nc0 + i;
        const int b  = n / HW;
        const int s  = n - b * HW;
        const int oh = s / WIMG;
        const int ow = s - oh * WIMG;
        xbase[i] = b * (CIN * HW) + oh * WIMG + ow;
        ohow[i]  = ((u32)oh << 16) | (u32)ow;
    }

    // Loop-invariant LDS staging destinations.
    uint4* dAh = (uint4*)(As_hi + tid * ASTR);
    uint4* dAl = (uint4*)(As_lo + tid * ASTR);
    u16*   dBh = Bs_hi + nc0 * BSTR + krow;
    u16*   dBl = Bs_lo + nc0 * BSTR + krow;

    // B prefetch registers only (8 VGPRs). PRESPLIT: packed bf16 pair;
    // fallback: raw f32 bits (split at store time).
    u32 rB[8];

    auto loadB = [&](int kk) {
        const int k  = kk + krow;
        const int ci = k / 9;
        const int r9 = k - ci * 9;
        const int kh = (r9 >= 6) ? 2 : (r9 >= 3 ? 1 : 0);
        const int kw = r9 - kh * 3;
        const int dh = kh - 1, dw = kw - 1;
        const int doff = ci * HW + dh * WIMG + dw;
#pragma unroll
        for (int i = 0; i < 8; ++i) {
            const int oh = (int)(ohow[i] >> 16);
            const int ow = (int)(ohow[i] & 0xFFFFu);
            const bool inb = ((u32)(oh + dh) < (u32)HIMG) &&
                             ((u32)(ow + dw) < (u32)WIMG);
            const int idx = xbase[i] + doff;
            if constexpr (PRESPLIT) rB[i] = inb ? xpk[idx] : 0u;
            else                    rB[i] = inb ? __float_as_uint(xf[idx]) : 0u;
        }
    };
    // A stage: global (near-cache after prefetch) -> short-lived regs -> LDS.
    auto stageA = [&](int kk) {
        if constexpr (PRESPLIT) {
            const uint4* sh = (const uint4*)(wh + (size_t)tid * KTOT + kk);
            const uint4* sl = (const uint4*)(wl + (size_t)tid * KTOT + kk);
            uint4 th[4], tl[4];
#pragma unroll
            for (int j = 0; j < 4; ++j) { th[j] = sh[j]; tl[j] = sl[j]; }
#pragma unroll
            for (int j = 0; j < 4; ++j) { dAh[j] = th[j]; dAl[j] = tl[j]; }
        } else {
            const uint4* s = (const uint4*)(wf + (size_t)tid * KTOT + kk);
            uint4 raw[8];
#pragma unroll
            for (int j = 0; j < 8; ++j) raw[j] = s[j];
            const u32* rf = (const u32*)raw;
            u32 hi[16], lo[16];
#pragma unroll
            for (int j = 0; j < 16; ++j) {
                const u32 p0 = split_pack(__uint_as_float(rf[2 * j]));
                const u32 p1 = split_pack(__uint_as_float(rf[2 * j + 1]));
                hi[j] = (p0 & 0xFFFFu) | (p1 << 16);
                lo[j] = (p0 >> 16) | (p1 & 0xFFFF0000u);
            }
            u32* dh32 = (u32*)dAh;
            u32* dl32 = (u32*)dAl;
#pragma unroll
            for (int j = 0; j < 16; ++j) { dh32[j] = hi[j]; dl32[j] = lo[j]; }
        }
    };
    auto stageB = [&]() {
#pragma unroll
        for (int i = 0; i < 8; ++i) {
            u32 pk;
            if constexpr (PRESPLIT) pk = rB[i];
            else                    pk = split_pack(__uint_as_float(rB[i]));
            dBh[i * BSTR] = (u16)pk;
            dBl[i * BSTR] = (u16)(pk >> 16);
        }
    };
    auto prefetchA = [&](int kk) {   // warm WGP$/L2 for next A tile, 0 VGPRs
        if constexpr (PRESPLIT) {
            __builtin_prefetch(wh + (size_t)tid * KTOT + kk, 0, 3);
            __builtin_prefetch(wl + (size_t)tid * KTOT + kk, 0, 3);
        } else {
            __builtin_prefetch(wf + (size_t)tid * KTOT + kk, 0, 3);
            __builtin_prefetch(wf + (size_t)tid * KTOT + kk + 16, 0, 3);
        }
    };

    // Accumulators with bias folded in. C/D layout: VGPR r holds
    // M = base + r + (lane<16 ? 0 : 8), N = base + lane%16.
    v8f acc[4][2];
#pragma unroll
    for (int tm = 0; tm < 4; ++tm) {
        const int co = waveM * 64 + tm * 16 + lhalf * 8;
        float bv[8];
#pragma unroll
        for (int r = 0; r < 8; ++r) bv[r] = bias[co + r];
#pragma unroll
        for (int tn = 0; tn < 2; ++tn)
#pragma unroll
            for (int r = 0; r < 8; ++r) acc[tm][tn][r] = bv[r];
    }

    // ---- software pipeline
    loadB(0);
    for (int kk = 0; kk < KTOT; kk += BK) {
        stageA(kk);
        stageB();
        __syncthreads();
        if (kk + BK < KTOT) {          // issue next gathers + A prefetch now;
            loadB(kk + BK);            // they complete under the WMMA block
            prefetchA(kk + BK);
        }

        v16bf bh[2], bl[2];
#pragma unroll
        for (int tn = 0; tn < 2; ++tn) {
            const int rn = waveN * 32 + tn * 16 + l16;
            bh[tn] = frag_B(Bs_hi + rn * BSTR, lhalf);
            bl[tn] = frag_B(Bs_lo + rn * BSTR, lhalf);
        }
#pragma unroll
        for (int tm = 0; tm < 4; ++tm) {
            const int rm = waveM * 64 + tm * 16 + l16;
            const v16bf ah = frag_A(As_hi + rm * ASTR, lhalf);
            const v16bf al = frag_A(As_lo + rm * ASTR, lhalf);
#pragma unroll
            for (int tn = 0; tn < 2; ++tn) {
                acc[tm][tn] = __builtin_amdgcn_wmma_f32_16x16x32_bf16(
                    false, ah, false, bh[tn], (short)0, acc[tm][tn], false, false);
                acc[tm][tn] = __builtin_amdgcn_wmma_f32_16x16x32_bf16(
                    false, al, false, bh[tn], (short)0, acc[tm][tn], false, false);
                acc[tm][tn] = __builtin_amdgcn_wmma_f32_16x16x32_bf16(
                    false, ah, false, bl[tn], (short)0, acc[tm][tn], false, false);
            }
        }
        __syncthreads();
    }

    // ---- Epilogue: NCHW store; 16 lanes = 16 consecutive ow -> coalesced.
#pragma unroll
    for (int tm = 0; tm < 4; ++tm) {
        const int co = waveM * 64 + tm * 16 + lhalf * 8;
#pragma unroll
        for (int tn = 0; tn < 2; ++tn) {
            const int n = n0 + waveN * 32 + tn * 16 + l16;
            const int b = n / HW;
            const int s = n - b * HW;
            float* dst = out + ((size_t)b * COUT + co) * HW + s;
#pragma unroll
            for (int r = 0; r < 8; ++r) dst[(size_t)r * HW] = acc[tm][tn][r];
        }
    }
}

extern "C" void kernel_launch(void* const* d_in, const int* in_sizes, int n_in,
                              void* d_out, int out_size, void* d_ws, size_t ws_size,
                              hipStream_t stream) {
    (void)in_sizes; (void)n_in; (void)out_size;
    const float* x    = (const float*)d_in[0];
    const float* w    = (const float*)d_in[1];
    const float* bias = (const float*)d_in[2];
    float* out        = (float*)d_out;

    // ws layout: xpk (NX dwords) | w_hi (NW halves) | w_lo (NW halves)
    const size_t need = (size_t)NX * 4 + (size_t)NW * 4;
    if (ws_size >= need) {
        u32* xpk = (u32*)d_ws;
        u16* wh  = (u16*)((char*)d_ws + (size_t)NX * 4);
        u16* wl  = wh + NW;
        splitpack_x<<<NX / 4 / 256, 256, 0, stream>>>((const float4*)x, (uint4*)xpk);
        split_w<<<NW / 256, 256, 0, stream>>>(w, wh, wl);
        conv_gemm<true><<<dim3(NTOT / BN), 256, 0, stream>>>(
            nullptr, nullptr, xpk, wh, wl, bias, out);
    } else {
        conv_gemm<false><<<dim3(NTOT / BN), 256, 0, stream>>>(
            x, w, nullptr, nullptr, nullptr, bias, out);
    }
}
